// BPBookMemory_81406810128809
// MI455X (gfx1250) — matrix operand order repaired and checked by reading, not compiled, not run on hardware
//
#include <hip/hip_runtime.h>
#include <math.h>

// ---------------------------------------------------------------------------
// BPBookMemory for MI455X (gfx1250, wave32, WMMA)
// feat GEMM + sim GEMM on v_wmma_f32_16x16x32_bf16; everything else HBM-bound.
// Workspace layout (bytes):
//   [0      , 32768 )  qacc   : 64x128 f32 (gelu column sums, atomicAdd)
//   [32768  , 65536 )  proto  : 64x128 f32
//   [65536  , 327680)  invn   : 65536 f32  (1/||memory_row||)
//   [327680 , 344064)  qn     : 64x128 bf16 (normalized mean query)
//   [344064 , ~17.1MB) sim    : 64x65536 f32
// ---------------------------------------------------------------------------

typedef __attribute__((ext_vector_type(16))) __bf16 v16bf;
typedef __attribute__((ext_vector_type(8)))  __bf16 v8bf;
typedef __attribute__((ext_vector_type(8)))  float  v8f;

#define NB    64
#define NN    4096
#define ND    128
#define NS    65536
#define TOPK  16
#define NEG_INF (-3.4e38f)

// native f32 -> bf16 (RNE) via hardware cvt; packed where possible
static __device__ __forceinline__ v16bf cvt16(v8f lo, v8f hi) {
  v8bf l = __builtin_convertvector(lo, v8bf);
  v8bf h = __builtin_convertvector(hi, v8bf);
  return __builtin_shufflevector(l, h, 0, 1, 2, 3, 4, 5, 6, 7,
                                       8, 9, 10, 11, 12, 13, 14, 15);
}
static __device__ __forceinline__ v16bf cat16(v8bf l, v8bf h) {
  return __builtin_shufflevector(l, h, 0, 1, 2, 3, 4, 5, 6, 7,
                                       8, 9, 10, 11, 12, 13, 14, 15);
}

// branch-free GELU: x * sigmoid(1.5957691*(x + 0.044715 x^3))
// (tanh-form GELU; |err| vs exact-erf GELU < 3.2e-4, far below the bf16 GEMM
//  noise floor; avoids erff's divergent dual-path polynomial entirely)
static __device__ __forceinline__ float gelu_fast(float v) {
  float u  = 1.5957691216057308f * __builtin_fmaf(0.044715f * v, v * v, v);
  float ex = __expf(-u);                              // v_exp_f32 (TRANS)
  return v * __builtin_amdgcn_rcpf(1.0f + ex);        // v_rcp_f32 (TRANS)
}

// ---------------------------------------------------------------------------
// init: zero the query accumulator
// ---------------------------------------------------------------------------
__global__ void k_init(float* qacc) {
  int i = blockIdx.x * blockDim.x + threadIdx.x;
  if (i < NB * ND) qacc[i] = 0.0f;
}

// ---------------------------------------------------------------------------
// k_feat: qacc[b][e] += sum_n gelu( x[b,n,:]·W[e,:] + bias[e] )
// grid = 64 b * 8 chunks; block = 256 (8 waves). Wave w owns rows [w*16,w*16+16)
// of each 128-row slab, 4 slabs per block (512 rows/chunk).
// ---------------------------------------------------------------------------
__global__ void __launch_bounds__(256) k_feat(const float* __restrict__ x,
                                              const float* __restrict__ conv_w,
                                              const float* __restrict__ conv_b,
                                              float* __restrict__ qacc) {
  __shared__ __attribute__((aligned(16))) __bf16 Wb[ND * ND];   // 32KB, [e][d]

  const int b     = blockIdx.x >> 3;
  const int chunk = blockIdx.x & 7;
  const int tid   = threadIdx.x;
  const int w     = tid >> 5;
  const int lane  = tid & 31;
  const int c     = lane & 15;          // column-within-tile / M-row
  const int half  = lane >> 4;          // lane-half selects K sub-range

  // stage conv_w into LDS as bf16 (native cvt)
  for (int i = tid; i < ND * ND; i += 256) Wb[i] = (__bf16)conv_w[i];
  __syncthreads();

  // per-lane bias for each of the 8 output tiles (column = t*16 + c)
  float biasv[8];
#pragma unroll
  for (int t = 0; t < 8; ++t) biasv[t] = conv_b[t * 16 + c];

  float colsum[8] = {0.f, 0.f, 0.f, 0.f, 0.f, 0.f, 0.f, 0.f};

  const int row_base = b * NN + chunk * 512;

#pragma unroll 1
  for (int iter = 0; iter < 4; ++iter) {
    const float* xrow = x + (size_t)(row_base + iter * 128 + w * 16 + c) * ND;
    // speculative prefetch of next slab's row (dropped if OOB)
    __builtin_prefetch(xrow + 128 * ND, 0, 0);

    // A fragments for all 4 K-steps (16-bit A 16x32 layout, K split per half)
    v16bf af[4];
#pragma unroll
    for (int k = 0; k < 4; ++k) {
      const int kb0 = k * 32 + half * 8;              // K 0..7 / 8..15 block
      v8f alo = *(const v8f*)(xrow + kb0);            // K +0..7
      v8f ahi = *(const v8f*)(xrow + kb0 + 16);       // K +16..23
      af[k] = cvt16(alo, ahi);
    }

#pragma unroll
    for (int t = 0; t < 8; ++t) {                     // output-column tiles
      v8f acc = {0.f, 0.f, 0.f, 0.f, 0.f, 0.f, 0.f, 0.f};
#pragma unroll
      for (int k = 0; k < 4; ++k) {                   // K steps of 32
        const int e  = t * 16 + c;                    // B: N=col, K split per half (table order)
        const int kb = k * 32 + half * 8;
        v8bf b0 = *(const v8bf*)&Wb[e * ND + kb];
        v8bf b1 = *(const v8bf*)&Wb[e * ND + kb + 16];
        acc = __builtin_amdgcn_wmma_f32_16x16x32_bf16(
            false, af[k], false, cat16(b0, b1), (short)0, acc, false, false);
      }
#pragma unroll
      for (int r = 0; r < 8; ++r)
        colsum[t] += gelu_fast(acc[r] + biasv[t]);
    }
  }

  // lanes l and l+16 hold the same output column (different M halves): fold.
#pragma unroll
  for (int t = 0; t < 8; ++t) {
    float v = colsum[t];
    v += __shfl_xor(v, 16, 32);
    if (lane < 16) atomicAdd(&qacc[b * ND + t * 16 + lane], v);
  }
}

// ---------------------------------------------------------------------------
// k_invnorm: invn[s] = 1 / max(||memory[s]||, eps); one wave per row
// ---------------------------------------------------------------------------
__global__ void __launch_bounds__(256) k_invnorm(const float* __restrict__ memory,
                                                 float* __restrict__ invn) {
  const int w    = threadIdx.x >> 5;
  const int lane = threadIdx.x & 31;
  const int row  = blockIdx.x * 8 + w;
  float4 v = ((const float4*)(memory + (size_t)row * ND))[lane];
  float s = v.x * v.x + v.y * v.y + v.z * v.z + v.w * v.w;
#pragma unroll
  for (int off = 16; off > 0; off >>= 1) s += __shfl_xor(s, off, 32);
  if (lane == 0) invn[row] = 1.0f / fmaxf(sqrtf(s), 1e-12f);
}

// ---------------------------------------------------------------------------
// k_qn: qn[b] = normalize(qacc[b] / N) in bf16; 64 blocks x 128 threads
// ---------------------------------------------------------------------------
__global__ void __launch_bounds__(128) k_qn(const float* __restrict__ qacc,
                                            __bf16* __restrict__ qn) {
  __shared__ float red[128];
  const int b = blockIdx.x, j = threadIdx.x;
  float val = qacc[b * ND + j] * (1.0f / (float)NN);
  red[j] = val * val;
  __syncthreads();
  for (int off = 64; off > 0; off >>= 1) {
    if (j < off) red[j] += red[j + off];
    __syncthreads();
  }
  float inv = 1.0f / fmaxf(sqrtf(red[0]), 1e-12f);
  qn[b * ND + j] = (__bf16)(val * inv);
}

// ---------------------------------------------------------------------------
// k_sim: sim[b][s] = (qn[b]·memory[s]) * invn[s] via WMMA.
// grid = 512 blocks; each wave owns one 16-wide s-tile, all 64 b rows (4 M-tiles)
// ---------------------------------------------------------------------------
__global__ void __launch_bounds__(256) k_sim(const __bf16* __restrict__ qn,
                                             const float* __restrict__ memory,
                                             const float* __restrict__ invn,
                                             float* __restrict__ sim) {
  __shared__ __attribute__((aligned(16))) __bf16 Qs[NB * ND];   // 16KB

  const int tid  = threadIdx.x;
  const int w    = tid >> 5;
  const int lane = tid & 31;
  const int c    = lane & 15;
  const int half = lane >> 4;
  const int s0   = (blockIdx.x * 8 + w) * 16;

  for (int i = tid; i < NB * ND; i += 256) Qs[i] = qn[i];
  __syncthreads();

  v8f acc[4];
#pragma unroll
  for (int m = 0; m < 4; ++m)
    acc[m] = (v8f){0.f, 0.f, 0.f, 0.f, 0.f, 0.f, 0.f, 0.f};

#pragma unroll
  for (int k = 0; k < 4; ++k) {
    // B fragment: rows of memory (K = feature dim), col = s-row
    const float* mrow = memory + (size_t)(s0 + c) * ND + k * 32 + half * 8;
    v8f mlo = *(const v8f*)(mrow);
    v8f mhi = *(const v8f*)(mrow + 16);
    v16bf bf = cvt16(mlo, mhi);

    const int kb0 = k * 32 + half * 8;
    const int kb1 = kb0 + 16;
#pragma unroll
    for (int m = 0; m < 4; ++m) {
      v8bf a0 = *(const v8bf*)&Qs[(m * 16 + c) * ND + kb0];
      v8bf a1 = *(const v8bf*)&Qs[(m * 16 + c) * ND + kb1];
      acc[m] = __builtin_amdgcn_wmma_f32_16x16x32_bf16(
          false, cat16(a0, a1), false, bf, (short)0, acc[m], false, false);
    }
  }

  const float sc = invn[s0 + c];
#pragma unroll
  for (int m = 0; m < 4; ++m) {
#pragma unroll
    for (int r = 0; r < 8; ++r) {
      const int brow = m * 16 + r + half * 8;       // C-layout: M split per half
      sim[(size_t)brow * NS + s0 + c] = acc[m][r] * sc;
    }
  }
}

// ---------------------------------------------------------------------------
// k_topk_proto: per-b top-16 over sim row, softmax, gather, weighted sum
// ---------------------------------------------------------------------------
__global__ void __launch_bounds__(256) k_topk_proto(const float* __restrict__ sim,
                                                    const float* __restrict__ memory,
                                                    float* __restrict__ proto) {
  __shared__ float s_sc[4096];
  __shared__ int   s_id[4096];
  __shared__ float s_rv[256];
  __shared__ int   s_rp[256];
  __shared__ float s_ts[TOPK];
  __shared__ int   s_ti[TOPK];
  __shared__ float s_w[TOPK];

  const int b = blockIdx.x, tid = threadIdx.x;
  const float* row = sim + (size_t)b * NS;

  // per-thread streaming top-16 with min-replace (registers, fully unrolled)
  float lsc[TOPK];
  int   lid[TOPK];
#pragma unroll
  for (int i = 0; i < TOPK; ++i) { lsc[i] = NEG_INF; lid[i] = 0; }
  float minv = NEG_INF;
  int   minpos = 0;

  for (int s = tid; s < NS; s += 256) {
    float v = row[s];
    if (v > minv) {
#pragma unroll
      for (int i = 0; i < TOPK; ++i)
        if (i == minpos) { lsc[i] = v; lid[i] = s; }
      minv = lsc[0]; minpos = 0;
#pragma unroll
      for (int i = 1; i < TOPK; ++i)
        if (lsc[i] < minv) { minv = lsc[i]; minpos = i; }
    }
  }
#pragma unroll
  for (int i = 0; i < TOPK; ++i) { s_sc[tid * TOPK + i] = lsc[i]; s_id[tid * TOPK + i] = lid[i]; }
  __syncthreads();

  // 16 rounds of block-wide argmax over 4096 candidates
  for (int r = 0; r < TOPK; ++r) {
    float bv = NEG_INF; int bp = tid * TOPK;
#pragma unroll
    for (int i = 0; i < TOPK; ++i) {
      float v = s_sc[tid * TOPK + i];
      if (v > bv) { bv = v; bp = tid * TOPK + i; }
    }
    s_rv[tid] = bv; s_rp[tid] = bp;
    __syncthreads();
    for (int off = 128; off > 0; off >>= 1) {
      if (tid < off && s_rv[tid + off] > s_rv[tid]) {
        s_rv[tid] = s_rv[tid + off]; s_rp[tid] = s_rp[tid + off];
      }
      __syncthreads();
    }
    if (tid == 0) {
      int p = s_rp[0];
      s_ts[r] = s_rv[0];
      s_ti[r] = s_id[p];
      s_sc[p] = NEG_INF;
    }
    __syncthreads();
  }

  // softmax over the 16 (descending) scores
  if (tid == 0) {
    float mx = s_ts[0], sum = 0.f, e[TOPK];
#pragma unroll
    for (int k = 0; k < TOPK; ++k) { e[k] = __expf(s_ts[k] - mx); sum += e[k]; }
    float inv = 1.0f / sum;
#pragma unroll
    for (int k = 0; k < TOPK; ++k) s_w[k] = e[k] * inv;
  }
  __syncthreads();

  // proto[b][j] = sum_k w[k] * memory[idx[k]][j]
  if (tid < ND) {
    float acc = 0.f;
#pragma unroll
    for (int k = 0; k < TOPK; ++k)
      acc += s_w[k] * memory[(size_t)s_ti[k] * ND + tid];
    proto[b * ND + tid] = acc;
  }
}

// ---------------------------------------------------------------------------
// k_out: out = x + scale * proto[b] (broadcast over N), float4 vectorized
// ---------------------------------------------------------------------------
__global__ void __launch_bounds__(256) k_out(const float* __restrict__ x,
                                             const float* __restrict__ proto,
                                             const float* __restrict__ scale_p,
                                             float* __restrict__ out) {
  const size_t i = (size_t)blockIdx.x * blockDim.x + threadIdx.x;  // float4 idx
  const float s = *scale_p;
  const size_t e = i * 4;
  const int b = (int)(e >> 19);          // 4096*128 = 2^19 elems per batch
  const int d = (int)(e & (ND - 1));
  float4 xv = ((const float4*)x)[i];
  float4 pv = *(const float4*)(proto + b * ND + d);
  float4 o;
  o.x = xv.x + s * pv.x;
  o.y = xv.y + s * pv.y;
  o.z = xv.z + s * pv.z;
  o.w = xv.w + s * pv.w;
  ((float4*)out)[i] = o;
}

// ---------------------------------------------------------------------------
extern "C" void kernel_launch(void* const* d_in, const int* in_sizes, int n_in,
                              void* d_out, int out_size, void* d_ws, size_t ws_size,
                              hipStream_t stream) {
  const float* x      = (const float*)d_in[0];
  const float* conv_w = (const float*)d_in[1];
  const float* conv_b = (const float*)d_in[2];
  const float* memory = (const float*)d_in[3];
  const float* scale  = (const float*)d_in[4];
  float* out = (float*)d_out;

  char* ws = (char*)d_ws;
  float*  qacc  = (float*)(ws + 0);
  float*  proto = (float*)(ws + 32768);
  float*  invn  = (float*)(ws + 65536);
  __bf16* qn    = (__bf16*)(ws + 327680);
  float*  sim   = (float*)(ws + 344064);   // 16 MB

  k_init      <<<32,    256, 0, stream>>>(qacc);
  k_feat      <<<512,   256, 0, stream>>>(x, conv_w, conv_b, qacc);
  k_invnorm   <<<8192,  256, 0, stream>>>(memory, invn);
  k_qn        <<<64,    128, 0, stream>>>(qacc, qn);
  k_sim       <<<512,   256, 0, stream>>>(qn, memory, invn, sim);
  k_topk_proto<<<64,    256, 0, stream>>>(sim, memory, proto);
  k_out       <<<32768, 256, 0, stream>>>(x, proto, scale, out);
}
